// TemporalWindowAttention_76544907149532
// MI455X (gfx1250) — matrix operand order repaired and checked by reading, not compiled, not run on hardware
//
#include <hip/hip_runtime.h>
#include <hip/hip_bf16.h>
#include <math.h>

// ---------------------------------------------------------------------------
// TemporalWindowAttention for MI455X (gfx1250, wave32):
//  - bf16 WMMA 16x16x32 for all matmuls (fp32 accumulate)
//  - TDM tensor_load_to_lds, double-buffered on TENSORcnt, for GEMM staging
//  - flash-style streaming softmax (never materializes the 512MB score tensor)
//
// hidden (16,320,32,32) fp32; F=16, C=320, heads=8, d=40, L=1024.
// Row ordering r = f*1024 + l everywhere.
// ---------------------------------------------------------------------------

typedef __bf16 bf16_t;
typedef __attribute__((ext_vector_type(16))) __bf16 v16bf;
typedef __attribute__((ext_vector_type(8)))  float  v8f;
typedef __attribute__((ext_vector_type(4)))  unsigned int u32x4;
typedef __attribute__((ext_vector_type(8)))  int  i32x8;
typedef __attribute__((ext_vector_type(4)))  int  i32x4;

#if __has_builtin(__builtin_amdgcn_tensor_load_to_lds) && \
    __has_builtin(__builtin_amdgcn_s_wait_tensorcnt)
#define USE_TDM 1
#else
#define USE_TDM 0
#endif

__device__ __forceinline__ v8f wmma_bf16(v16bf a, v16bf b, v8f c) {
  return __builtin_amdgcn_wmma_f32_16x16x32_bf16(
      false, a, false, b, (short)0, c, false, false);
}

constexpr int kC  = 320;
constexpr int kF  = 16;
constexpr int kHd = 8;
constexpr int kDH = 40;
constexpr int kDP = 64;
constexpr int kL  = 1024;
constexpr int kFH = kF * kHd;  // 128
constexpr int kKS = kC / 32;   // 10 K-steps

#if USE_TDM
// ---------------------------------------------------------------------------
// One TDM 2D tile load: tile_d1 rows x tile_d0 contiguous fp32 elements
// (row stride = stride_elems), packed row-major into LDS at lds_byte_off.
// D# per CDNA5 ISA ch.8 (group0 128b, group1 256b; groups 2/3 zeroed).
// ---------------------------------------------------------------------------
__device__ __forceinline__ void tdm_load_2d_f32(unsigned lds_byte_off,
                                                const void* gptr,
                                                unsigned tile_d0, unsigned tile_d1,
                                                unsigned stride_elems) {
  unsigned long long ga = (unsigned long long)(uintptr_t)gptr;
  u32x4 g0;
  g0[0] = 1u;                                    // count=1 (valid), user mode
  g0[1] = lds_byte_off;                          // lds_addr[31:0]
  g0[2] = (unsigned)(ga & 0xFFFFFFFFu);          // global_addr[31:0]
  g0[3] = (unsigned)((ga >> 32) & 0x01FFFFFFu)   // global_addr[56:32]
        | 0x80000000u;                           // type=2 ("image") in [127:126]

  const unsigned td0 = tile_d0, td1 = tile_d1;   // tensor dims = tile (interior)
  i32x8 g1;
  g1[0] = (int)(2u << 16);                       // data_size=2 (4B); mask=0
  g1[1] = (int)((td0 & 0xFFFFu) << 16);          // tensor_dim0[15:0] @ bit48
  g1[2] = (int)(((td0 >> 16) & 0xFFFFu) | ((td1 & 0xFFFFu) << 16));
  g1[3] = (int)(((td1 >> 16) & 0xFFFFu) | ((tile_d0 & 0xFFFFu) << 16)); // tile_dim0
  g1[4] = (int)(tile_d1 & 0xFFFFu);              // tile_dim1; tile_dim2=0
  g1[5] = (int)stride_elems;                     // tensor_dim0_stride[31:0]
  g1[6] = 0;
  g1[7] = 0;

  i32x4 z4; z4[0] = 0; z4[1] = 0; z4[2] = 0; z4[3] = 0;
#if __clang_major__ >= 23
  i32x8 z8; z8[0]=0; z8[1]=0; z8[2]=0; z8[3]=0; z8[4]=0; z8[5]=0; z8[6]=0; z8[7]=0;
  __builtin_amdgcn_tensor_load_to_lds(g0, g1, z4, z4, z8, 0);
#else
  __builtin_amdgcn_tensor_load_to_lds(g0, g1, z4, z4, 0);
#endif
}
#endif  // USE_TDM

// ---- shared WMMA K-step helpers (fp32 LDS tiles -> bf16 operands) ----------
// A tile stored [k][m] (64 m contiguous), B tile stored [n][k] (32 k contiguous)
__device__ __forceinline__ void gemm_step_Akm(const float (*__restrict__ A)[64],
                                              const float (*__restrict__ B)[32],
                                              int m, int hi, int lo, v8f acc[4]) {
  v16bf a;
#pragma unroll
  for (int e = 0; e < 8; ++e) {
    a[e]     = (bf16_t)A[hi * 16 + e][m];
    a[e + 8] = (bf16_t)A[hi * 16 + 8 + e][m];
  }
#pragma unroll
  for (int g = 0; g < 4; ++g) {
    const float* brow = &B[g * 16 + lo][hi * 16];
    v16bf b;
#pragma unroll
    for (int e = 0; e < 16; ++e) b[e] = (bf16_t)brow[e];
    acc[g] = wmma_bf16(a, b, acc[g]);
  }
}
// A tile stored [m][k] (32 k contiguous), B tile stored [n][k]
__device__ __forceinline__ void gemm_step_Amk(const float (*__restrict__ A)[32],
                                              const float (*__restrict__ B)[32],
                                              int m, int hi, int lo, v8f acc[4]) {
  v16bf a;
#pragma unroll
  for (int e = 0; e < 8; ++e) {
    a[e]     = (bf16_t)A[m][hi * 16 + e];
    a[e + 8] = (bf16_t)A[m][hi * 16 + 8 + e];
  }
#pragma unroll
  for (int g = 0; g < 4; ++g) {
    const float* brow = &B[g * 16 + lo][hi * 16];
    v16bf b;
#pragma unroll
    for (int e = 0; e < 16; ++e) b[e] = (bf16_t)brow[e];
    acc[g] = wmma_bf16(a, b, acc[g]);
  }
}

// ---------------------------------------------------------------------------
// Kernel 1: QKV projection. GEMM M=16384 (r=f*1024+l), N=960, K=320.
// Double-buffered TDM staging; epilogue: +bias, Q*1/sqrt(40); scatter to
// padded Q,K [fh][l][64] and transposed V [fh][64][l] (bf16).
// grid (256, 15), block 128 (4 waves).
// ---------------------------------------------------------------------------
__global__ __launch_bounds__(128)
void qkv_proj_kernel(const float* __restrict__ hs, const float* __restrict__ win,
                     const float* __restrict__ bin,
                     bf16_t* __restrict__ Qp, bf16_t* __restrict__ Kp,
                     bf16_t* __restrict__ Vt) {
  __shared__ __align__(16) float Ald[2][32][64];  // [k][m] packed (TDM layout)
  __shared__ __align__(16) float Bld[2][64][32];  // [n][k] packed

  const int t    = threadIdx.x;
  const int lane = t & 31;
  const int wv   = t >> 5;
  const int lo   = lane & 15;
  const int hi   = lane >> 4;

  const int r0 = blockIdx.x * 64;
  const int f  = r0 >> 10;
  const int l0 = r0 & 1023;
  const int n0 = blockIdx.y * 64;
  const int m  = wv * 16 + lo;

  const float* gA = hs + (size_t)f * kC * kL + l0;   // + k0*kL
  const float* gB = win + (size_t)n0 * kC;           // + k0

  v8f acc[4] = {};

#if USE_TDM
  if (t < 32) {  // wave 0 drives the DMA pipeline (TENSORcnt-tracked, in-order)
    tdm_load_2d_f32((unsigned)(uintptr_t)&Ald[0][0][0], gA, 64, 32, kL);
    tdm_load_2d_f32((unsigned)(uintptr_t)&Bld[0][0][0], gB, 32, 64, kC);
  }
  for (int ks = 0; ks < kKS - 1; ++ks) {
    const int cur = ks & 1;
    if (t < 32) {  // prefetch step ks+1 into the other buffer
      const int k1 = (ks + 1) * 32;
      tdm_load_2d_f32((unsigned)(uintptr_t)&Ald[cur ^ 1][0][0], gA + (size_t)k1 * kL,
                      64, 32, kL);
      tdm_load_2d_f32((unsigned)(uintptr_t)&Bld[cur ^ 1][0][0], gB + k1, 32, 64, kC);
    }
    // <=2 outstanding allowed (the pair just issued); step-ks pair has landed.
    __builtin_amdgcn_s_wait_tensorcnt(2);
    __syncthreads();
    gemm_step_Akm(Ald[cur], Bld[cur], m, hi, lo, acc);
    __syncthreads();  // buffer 'cur' is re-filled next iteration
  }
  __builtin_amdgcn_s_wait_tensorcnt(0);
  __syncthreads();
  gemm_step_Akm(Ald[(kKS - 1) & 1], Bld[(kKS - 1) & 1], m, hi, lo, acc);
#else
  for (int k0 = 0; k0 < kC; k0 += 32) {
#pragma unroll
    for (int i = 0; i < 4; ++i) {
      int idx = i * 128 + t, kc = idx >> 4, m4 = (idx & 15) * 4;
      *(float4*)&Ald[0][kc][m4] = *(const float4*)&gA[(size_t)(k0 + kc) * kL + m4];
    }
#pragma unroll
    for (int i = 0; i < 4; ++i) {
      int idx = i * 128 + t, nc = idx >> 3, k4 = (idx & 7) * 4;
      *(float4*)&Bld[0][nc][k4] = *(const float4*)&gB[(size_t)nc * kC + k0 + k4];
    }
    __syncthreads();
    gemm_step_Akm(Ald[0], Bld[0], m, hi, lo, acc);
    __syncthreads();
  }
#endif

  const float qscale = 0.15811388300841898f;  // 1/sqrt(40)
#pragma unroll
  for (int g = 0; g < 4; ++g) {
    const int n  = n0 + g * 16 + lo;
    const float bv = bin[n];
#pragma unroll
    for (int i = 0; i < 8; ++i) {
      const int l = l0 + wv * 16 + i + hi * 8;
      const float v = acc[g][i] + bv;
      if (n < kC) {
        int h = n / kDH, dd = n % kDH;
        Qp[((size_t)(f * kHd + h) * kL + l) * kDP + dd] = (bf16_t)(v * qscale);
      } else if (n < 2 * kC) {
        int nn = n - kC; int h = nn / kDH, dd = nn % kDH;
        Kp[((size_t)(f * kHd + h) * kL + l) * kDP + dd] = (bf16_t)v;
      } else {
        int nn = n - 2 * kC; int h = nn / kDH, dd = nn % kDH;
        Vt[((size_t)(f * kHd + h) * kDP + dd) * kL + l] = (bf16_t)v;  // transposed
      }
    }
  }
}

// ---------------------------------------------------------------------------
// Kernel 2: flash attention. One wave per (f*h, 16 query rows); streams keys in
// 32-col steps: 4 WMMAs for S (K=64 incl zero pad), online softmax via 2KB LDS
// tile, 3 WMMAs for P*V. K/V are L2-resident (50MB << 192MB L2).
// grid (128, 64), block 32.
// ---------------------------------------------------------------------------
__global__ __launch_bounds__(32)
void flash_attn_kernel(const bf16_t* __restrict__ Qp, const bf16_t* __restrict__ Kp,
                       const bf16_t* __restrict__ Vt, float* __restrict__ Obuf) {
  __shared__ __align__(32) float  Sld[16][32];
  __shared__ __align__(32) bf16_t Pld[16][32];
  __shared__ float alphald[16];
  __shared__ float rowscale[16];

  const int lane = threadIdx.x;
  const int lo   = lane & 15;
  const int hi   = lane >> 4;
  const int fh   = blockIdx.x;
  const int f    = fh >> 3;
  const int h    = fh & 7;
  const int l0   = blockIdx.y * 16;

  const bf16_t* Qb = Qp + (size_t)fh * kL * kDP;
  const bf16_t* Kb = Kp + (size_t)fh * kL * kDP;
  const bf16_t* Vb = Vt + (size_t)fh * kDP * kL;

  v16bf aq0, aq1;
  {
    const bf16_t* qrow = Qb + (size_t)(l0 + lo) * kDP;
#pragma unroll
    for (int e = 0; e < 8; ++e) {
      aq0[e]     = qrow[hi * 16 + e];
      aq0[e + 8] = qrow[hi * 16 + 8 + e];
      aq1[e]     = qrow[32 + hi * 16 + e];
      aq1[e + 8] = qrow[32 + hi * 16 + 8 + e];
    }
  }

  v8f o0 = {}, o1 = {}, o2 = {};
  float m_r = -3.0e38f, l_r = 0.f;

  for (int mb = 0; mb < kL; mb += 32) {
    v8f s0, s1;
#pragma unroll
    for (int g = 0; g < 2; ++g) {
      const bf16_t* krow = Kb + (size_t)(mb + g * 16 + lo) * kDP;
      v16bf b0 = *(const v16bf*)(krow + hi * 16);        // dd 0..31
      v16bf b1 = *(const v16bf*)(krow + 32 + hi * 16);   // dd 32..63
      v8f c = {};
      c = wmma_bf16(aq0, b0, c);
      c = wmma_bf16(aq1, b1, c);
      if (g == 0) s0 = c; else s1 = c;
    }
#pragma unroll
    for (int i = 0; i < 8; ++i) {
      Sld[i + hi * 8][lo]      = s0[i];
      Sld[i + hi * 8][16 + lo] = s1[i];
    }
    __syncthreads();

    if (lane < 16) {  // one row per lane
      float mt = -3.0e38f;
#pragma unroll
      for (int c = 0; c < 32; ++c) mt = fmaxf(mt, Sld[lane][c]);
      const float mnew  = fmaxf(m_r, mt);
      const float alpha = __expf(m_r - mnew);
      float sum = 0.f;
#pragma unroll
      for (int c = 0; c < 32; ++c) {
        const float p = __expf(Sld[lane][c] - mnew);
        sum += p;
        Pld[lane][c] = (bf16_t)p;
      }
      l_r = l_r * alpha + sum;
      m_r = mnew;
      alphald[lane] = alpha;
    }
    __syncthreads();

    float al[8];
#pragma unroll
    for (int i = 0; i < 8; ++i) al[i] = alphald[i + hi * 8];
#pragma unroll
    for (int i = 0; i < 8; ++i) { o0[i] *= al[i]; o1[i] *= al[i]; o2[i] *= al[i]; }

    v16bf ap;
#pragma unroll
    for (int e = 0; e < 8; ++e) {
      ap[e]     = Pld[lo][hi * 16 + e];
      ap[e + 8] = Pld[lo][hi * 16 + 8 + e];
    }
#pragma unroll
    for (int ng = 0; ng < 3; ++ng) {
      v16bf b = *(const v16bf*)(Vb + (size_t)(ng * 16 + lo) * kL + mb + hi * 16);
      if (ng == 0)      o0 = wmma_bf16(ap, b, o0);
      else if (ng == 1) o1 = wmma_bf16(ap, b, o1);
      else              o2 = wmma_bf16(ap, b, o2);
    }
    __syncthreads();
  }

  if (lane < 16) rowscale[lane] = (l_r > 0.f) ? (1.f / l_r) : 0.f;
  __syncthreads();
  float rs[8];
#pragma unroll
  for (int i = 0; i < 8; ++i) rs[i] = rowscale[i + hi * 8];

#pragma unroll
  for (int ng = 0; ng < 3; ++ng) {
    const int col = ng * 16 + lo;
    if (col < kDH) {
#pragma unroll
      for (int i = 0; i < 8; ++i) {
        const int l = l0 + i + hi * 8;
        const float v = (ng == 0 ? o0[i] : (ng == 1 ? o1[i] : o2[i])) * rs[i];
        Obuf[((size_t)f * kL + l) * kC + h * kDH + col] = v;
      }
    }
  }
}

// ---------------------------------------------------------------------------
// Kernel 3: out projection + bias + residual. GEMM M=16384, N=320, K=320.
// Double-buffered TDM staging. grid (256, 5), block 128.
// ---------------------------------------------------------------------------
__global__ __launch_bounds__(128)
void out_proj_kernel(const float* __restrict__ Obuf, const float* __restrict__ wout,
                     const float* __restrict__ bout, const float* __restrict__ hs,
                     float* __restrict__ out) {
  __shared__ __align__(16) float Ald[2][64][32];  // [m][k] packed
  __shared__ __align__(16) float Bld[2][64][32];  // [n][k] packed

  const int t    = threadIdx.x;
  const int lane = t & 31;
  const int wv   = t >> 5;
  const int lo   = lane & 15;
  const int hi   = lane >> 4;

  const int r0 = blockIdx.x * 64;
  const int f  = r0 >> 10;
  const int l0 = r0 & 1023;
  const int n0 = blockIdx.y * 64;
  const int m  = wv * 16 + lo;

  const float* gA = Obuf + (size_t)r0 * kC;   // + k0
  const float* gB = wout + (size_t)n0 * kC;   // + k0

  v8f acc[4] = {};

#if USE_TDM
  if (t < 32) {
    tdm_load_2d_f32((unsigned)(uintptr_t)&Ald[0][0][0], gA, 32, 64, kC);
    tdm_load_2d_f32((unsigned)(uintptr_t)&Bld[0][0][0], gB, 32, 64, kC);
  }
  for (int ks = 0; ks < kKS - 1; ++ks) {
    const int cur = ks & 1;
    if (t < 32) {
      const int k1 = (ks + 1) * 32;
      tdm_load_2d_f32((unsigned)(uintptr_t)&Ald[cur ^ 1][0][0], gA + k1, 32, 64, kC);
      tdm_load_2d_f32((unsigned)(uintptr_t)&Bld[cur ^ 1][0][0], gB + k1, 32, 64, kC);
    }
    __builtin_amdgcn_s_wait_tensorcnt(2);
    __syncthreads();
    gemm_step_Amk(Ald[cur], Bld[cur], m, hi, lo, acc);
    __syncthreads();
  }
  __builtin_amdgcn_s_wait_tensorcnt(0);
  __syncthreads();
  gemm_step_Amk(Ald[(kKS - 1) & 1], Bld[(kKS - 1) & 1], m, hi, lo, acc);
#else
  for (int k0 = 0; k0 < kC; k0 += 32) {
#pragma unroll
    for (int i = 0; i < 4; ++i) {
      int idx = i * 128 + t, mr = idx >> 3, k4 = (idx & 7) * 4;
      *(float4*)&Ald[0][mr][k4] = *(const float4*)&gA[(size_t)mr * kC + k0 + k4];
    }
#pragma unroll
    for (int i = 0; i < 4; ++i) {
      int idx = i * 128 + t, nc = idx >> 3, k4 = (idx & 7) * 4;
      *(float4*)&Bld[0][nc][k4] = *(const float4*)&gB[(size_t)nc * kC + k0 + k4];
    }
    __syncthreads();
    gemm_step_Amk(Ald[0], Bld[0], m, hi, lo, acc);
    __syncthreads();
  }
#endif

#pragma unroll
  for (int g = 0; g < 4; ++g) {
    const int n  = n0 + g * 16 + lo;
    const float bv = bout[n];
#pragma unroll
    for (int i = 0; i < 8; ++i) {
      const int l = l0 + wv * 16 + i + hi * 8;
      const size_t idx = ((size_t)f * kC + n) * kL + l;
      out[idx] = acc[g][i] + bv + hs[idx];
    }
  }
}

// ---------------------------------------------------------------------------
// Launcher
// ---------------------------------------------------------------------------
extern "C" void kernel_launch(void* const* d_in, const int* in_sizes, int n_in,
                              void* d_out, int out_size, void* d_ws, size_t ws_size,
                              hipStream_t stream) {
  const float* hs   = (const float*)d_in[0];
  const float* wIn  = (const float*)d_in[1];
  const float* bIn  = (const float*)d_in[2];
  const float* wOut = (const float*)d_in[3];
  const float* bOut = (const float*)d_in[4];
  float* out = (float*)d_out;

  const size_t qBytes = (size_t)kFH * kL * kDP * sizeof(bf16_t);  // 16.78 MB
  const size_t oBytes = (size_t)kF * kL * kC * sizeof(float);     // 20.97 MB
  const size_t need   = 3 * qBytes + oBytes;                      // ~71.3 MB
  if (ws_size < need) return;

  char* ws = (char*)d_ws;
  bf16_t* Qp   = (bf16_t*)(ws);
  bf16_t* Kp   = (bf16_t*)(ws + qBytes);
  bf16_t* Vt   = (bf16_t*)(ws + 2 * qBytes);
  float*  Obuf = (float*)(ws + 3 * qBytes);

  hipMemsetAsync(ws, 0, 3 * qBytes, stream);  // zero d=40->64 pad lanes

  qkv_proj_kernel<<<dim3(256, 15), 128, 0, stream>>>(hs, wIn, bIn, Qp, Kp, Vt);
  flash_attn_kernel<<<dim3(kFH, kL / 16), 32, 0, stream>>>(Qp, Kp, Vt, Obuf);
  out_proj_kernel<<<dim3(256, 5), 128, 0, stream>>>(Obuf, wOut, bOut, hs, out);
}
